// selfattention_40295383171639
// MI455X (gfx1250) — matrix operand order repaired and checked by reading, not compiled, not run on hardware
//
#include <hip/hip_runtime.h>
#include <hip/hip_bf16.h>
#include <stdint.h>

// MI455X (gfx1250) attention (no softmax) — bf16 WMMA + async LDS pipeline.
// ~481 GFLOP, compute-bound: all matmuls via v_wmma_f32_16x16x32_bf16 with
// double-buffered GLOBAL_LOAD_ASYNC_TO_LDS_B128 staging (ASYNCcnt), K-tile 64
// (16 WMMAs per barrier pair), statically unrolled ping-pong buffers.
// Workspace: xb(64M) Wq/Wk/Wv_b(6M) Q/K/V/Vt(256M) Scores(128M) ~= 454MB.

typedef __bf16 bf16;
typedef __attribute__((ext_vector_type(16))) __bf16 v16bf;
typedef __attribute__((ext_vector_type(8)))  __bf16 v8bf;
typedef __attribute__((ext_vector_type(8)))  float  v8f;
typedef int v4i __attribute__((vector_size(16)));

#define TILE_M 128
#define TILE_N 128
#define TILE_K 64            // staged K per buffer (two k=32 WMMA sub-steps)

// ---- async 16-byte global->LDS copy (CDNA5, tracked by ASYNCcnt) ----------
__device__ __forceinline__ void async_b128(const bf16* g, bf16* l) {
  typedef __attribute__((address_space(1))) v4i* gas_t;   // global v4i*
  typedef __attribute__((address_space(3))) v4i* las_t;   // LDS v4i*
  __builtin_amdgcn_global_load_async_to_lds_b128(
      (gas_t)(void*)g, (las_t)(void*)l, 0, 0);
}

__device__ __forceinline__ void wait_async_le8() {
  asm volatile("s_wait_asynccnt 0x8" ::: "memory");
}
__device__ __forceinline__ void wait_async_le0() {
  asm volatile("s_wait_asynccnt 0x0" ::: "memory");
}

// Build a WMMA fragment (v16bf) from two 16-byte LDS chunks.
__device__ __forceinline__ v16bf frag2(const bf16* s, int off0, int off1) {
  v8bf a0 = *(const v8bf*)(s + off0);
  v8bf a1 = *(const v8bf*)(s + off1);
  v16bf r;
#pragma unroll
  for (int i = 0; i < 8; ++i) { r[i] = a0[i]; r[i + 8] = a1[i]; }
  return r;
}

// ---------------------------------------------------------------------------
// NT GEMM, bf16 in: C[m,n] = (sum_k A[m,k]*B[n,k] + bias[n]) * alpha.
// lda = ldb = K, ldc = N. Batched over blockIdx.z. Double-buffered async LDS.
// Requires K % 128 == 0 (nk = K/64 even, >= 2): holds for K in {1024, 2048}.
// ---------------------------------------------------------------------------
template <typename OutT>
__global__ __launch_bounds__(256, 2)
void gemm_nt_async(const bf16* __restrict__ A, const bf16* __restrict__ B,
                   OutT* __restrict__ C, const float* __restrict__ bias,
                   float alpha, int M, int N, int K,
                   long long strideA, long long strideB, long long strideC)
{
  __shared__ __align__(32) bf16 shA[2][TILE_M * TILE_K];  // 2 x 16KB
  __shared__ __align__(32) bf16 shB[2][TILE_N * TILE_K];  // 2 x 16KB

  A += (size_t)blockIdx.z * strideA;
  B += (size_t)blockIdx.z * strideB;
  C += (size_t)blockIdx.z * strideC;

  const int nBase = blockIdx.x * TILE_N;
  const int mBase = blockIdx.y * TILE_M;

  const int t    = threadIdx.x;
  const int lrow = t >> 1;            // 0..127 staging row
  const int lkc  = (t & 1) * 32;      // 0 / 32 staging k-chunk (32 elems = 64B)

  const int wave = t >> 5;
  const int lane = t & 31;
  const int wm = (wave >> 1) * 32;    // wave tile: 32 rows
  const int wn = (wave & 1) * 64;     // wave tile: 64 cols
  const int lh = lane >> 4;           // lane half
  const int lr = lane & 15;           // lane row (A) / col (B, C)

  const bf16* gArow = A + (size_t)(mBase + lrow) * K + lkc;
  const bf16* gBrow = B + (size_t)(nBase + lrow) * K + lkc;
  bf16* lA[2] = { &shA[0][lrow * TILE_K + lkc], &shA[1][lrow * TILE_K + lkc] };
  bf16* lB[2] = { &shB[0][lrow * TILE_K + lkc], &shB[1][lrow * TILE_K + lkc] };

  v8f acc[2][4] = {};

  // 8 async B128 copies per staged tile pair (4 for A, 4 for B).
  auto stage = [&](int buf, int k0) {
#pragma unroll
    for (int c = 0; c < 4; ++c) async_b128(gArow + k0 + c * 8, lA[buf] + c * 8);
#pragma unroll
    for (int c = 0; c < 4; ++c) async_b128(gBrow + k0 + c * 8, lB[buf] + c * 8);
  };

  // 16 WMMAs on one staged buffer (two k=32 sub-steps).
  auto compute = [&](const bf16* sA, const bf16* sB) {
#pragma unroll
    for (int kk = 0; kk < TILE_K; kk += 32) {
      v16bf af[2], bfr[4];
#pragma unroll
      for (int mi = 0; mi < 2; ++mi) {
        int r = (wm + mi * 16 + lr) * TILE_K + kk;
        af[mi] = frag2(sA, r + lh * 8, r + 16 + lh * 8);
      }
#pragma unroll
      for (int ni = 0; ni < 4; ++ni) {
        int r = (wn + ni * 16 + lr) * TILE_K + kk;
        bfr[ni] = frag2(sB, r + lh * 8, r + 16 + lh * 8);
      }
#pragma unroll
      for (int mi = 0; mi < 2; ++mi)
#pragma unroll
        for (int ni = 0; ni < 4; ++ni)
          acc[mi][ni] = __builtin_amdgcn_wmma_f32_16x16x32_bf16(
              false, af[mi], false, bfr[ni], (short)0, acc[mi][ni], false, false);
    }
  };

  const int nk = K / TILE_K;          // even, >= 2
  stage(0, 0);

  for (int i = 0; i + 2 < nk; i += 2) {
    stage(1, (i + 1) * TILE_K);
    wait_async_le8();                 // buffer 0 copies complete (in-order)
    __syncthreads();
    compute(shA[0], shB[0]);
    __syncthreads();                  // buffer 0 free for next stage

    stage(0, (i + 2) * TILE_K);
    wait_async_le8();                 // buffer 1 copies complete
    __syncthreads();
    compute(shA[1], shB[1]);
    __syncthreads();                  // buffer 1 free for next stage
  }
  // Tail: tiles nk-2 (buf 0) and nk-1 (buf 1).
  stage(1, (nk - 1) * TILE_K);
  wait_async_le8();
  __syncthreads();
  compute(shA[0], shB[0]);
  wait_async_le0();
  __syncthreads();
  compute(shA[1], shB[1]);

#pragma unroll
  for (int mi = 0; mi < 2; ++mi)
#pragma unroll
    for (int ni = 0; ni < 4; ++ni) {
      const int col = nBase + wn + ni * 16 + lr;
      const float b = bias ? bias[col] : 0.0f;
#pragma unroll
      for (int i = 0; i < 8; ++i) {
        const int row = mBase + wm + mi * 16 + lh * 8 + i;
        C[(size_t)row * N + col] = (OutT)((acc[mi][ni][i] + b) * alpha);
      }
    }
}

// ---------------------------------------------------------------------------
// f32 -> bf16 elementwise convert (8 elements/thread).
// ---------------------------------------------------------------------------
__global__ __launch_bounds__(256)
void f32_to_bf16(const float* __restrict__ in, bf16* __restrict__ out,
                 long long n)
{
  long long i = ((long long)blockIdx.x * 256 + threadIdx.x) * 8;
  if (i + 8 > n) return;
  float4 a = *(const float4*)(in + i);
  float4 b = *(const float4*)(in + i + 4);
  bf16 o[8] = {(bf16)a.x, (bf16)a.y, (bf16)a.z, (bf16)a.w,
               (bf16)b.x, (bf16)b.y, (bf16)b.z, (bf16)b.w};
  *(uint4*)(out + i) = *(const uint4*)o;
}

// ---------------------------------------------------------------------------
// Per-head bf16 transpose: out[c][r] = in[r][c], batched over blockIdx.z.
// ---------------------------------------------------------------------------
__global__ __launch_bounds__(256)
void transpose_bf16(const bf16* __restrict__ in, bf16* __restrict__ out,
                    int rows, int cols, long long strideIn, long long strideOut)
{
  __shared__ bf16 tile[32][33];
  in  += (size_t)blockIdx.z * strideIn;
  out += (size_t)blockIdx.z * strideOut;

  const int tx = threadIdx.x & 31;
  const int ty = threadIdx.x >> 5;   // 0..7
  const int c0 = blockIdx.x * 32;
  const int r0 = blockIdx.y * 32;

#pragma unroll
  for (int j = 0; j < 32; j += 8)
    tile[ty + j][tx] = in[(size_t)(r0 + ty + j) * cols + c0 + tx];
  __syncthreads();
#pragma unroll
  for (int j = 0; j < 32; j += 8)
    out[(size_t)(c0 + ty + j) * rows + r0 + tx] = tile[tx][ty + j];
}

// ---------------------------------------------------------------------------
extern "C" void kernel_launch(void* const* d_in, const int* in_sizes, int n_in,
                              void* d_out, int out_size, void* d_ws, size_t ws_size,
                              hipStream_t stream) {
  const float* x  = (const float*)d_in[0];
  const float* Wq = (const float*)d_in[1];
  const float* bq = (const float*)d_in[2];
  const float* Wk = (const float*)d_in[3];
  const float* bk = (const float*)d_in[4];
  const float* Wv = (const float*)d_in[5];
  const float* bv = (const float*)d_in[6];
  float* out = (float*)d_out;

  const int BH = 16;                 // B*H
  const int S  = 2048;
  const int D  = 1024;
  const int M  = BH * S;             // 32768
  const float inv_sqrt_d = 0.03125f; // 1/sqrt(1024)

  bf16* xb  = (bf16*)d_ws;
  bf16* Wqb = xb  + (size_t)M * D;
  bf16* Wkb = Wqb + (size_t)D * D;
  bf16* Wvb = Wkb + (size_t)D * D;
  bf16* Q   = Wvb + (size_t)D * D;
  bf16* Km  = Q   + (size_t)M * D;
  bf16* V   = Km  + (size_t)M * D;
  bf16* Vt  = V   + (size_t)M * D;
  bf16* Sc  = Vt  + (size_t)M * D;   // [BH][S][S]

  dim3 blk(256);

  // 0) bf16 conversions (bandwidth-trivial vs. the matmul FLOPs).
  f32_to_bf16<<<dim3(((long long)M * D) / (256 * 8)), blk, 0, stream>>>(x, xb, (long long)M * D);
  f32_to_bf16<<<dim3(((long long)D * D) / (256 * 8)), blk, 0, stream>>>(Wq, Wqb, (long long)D * D);
  f32_to_bf16<<<dim3(((long long)D * D) / (256 * 8)), blk, 0, stream>>>(Wk, Wkb, (long long)D * D);
  f32_to_bf16<<<dim3(((long long)D * D) / (256 * 8)), blk, 0, stream>>>(Wv, Wvb, (long long)D * D);

  // 1) QKV projections (scale folded into Q).
  dim3 g1(D / TILE_N, M / TILE_M, 1);
  gemm_nt_async<bf16><<<g1, blk, 0, stream>>>(xb, Wqb, Q,  bq, inv_sqrt_d, M, D, D, 0, 0, 0);
  gemm_nt_async<bf16><<<g1, blk, 0, stream>>>(xb, Wkb, Km, bk, 1.0f,       M, D, D, 0, 0, 0);
  gemm_nt_async<bf16><<<g1, blk, 0, stream>>>(xb, Wvb, V,  bv, 1.0f,       M, D, D, 0, 0, 0);

  // 2) V -> V^T per head so the last GEMM is NT too.
  dim3 gt(D / 32, S / 32, BH);
  transpose_bf16<<<gt, blk, 0, stream>>>(V, Vt, S, D, (long long)S * D, (long long)S * D);

  // 3) Scores = Q * K^T (scale already in Q), bf16 out.
  dim3 g2(S / TILE_N, S / TILE_M, BH);
  gemm_nt_async<bf16><<<g2, blk, 0, stream>>>(
      Q, Km, Sc, nullptr, 1.0f, S, S, D,
      (long long)S * D, (long long)S * D, (long long)S * S);

  // 4) Out = Scores * V^T(NT), f32 straight to d_out.
  dim3 g3(D / TILE_N, S / TILE_M, BH);
  gemm_nt_async<float><<<g3, blk, 0, stream>>>(
      Sc, Vt, out, nullptr, 1.0f, S, D, S,
      (long long)S * S, (long long)D * S, (long long)S * D);
}
